// Net_49976239456390
// MI455X (gfx1250) — matrix operand order repaired and checked by reading, not compiled, not run on hardware
//
#include <hip/hip_runtime.h>
#include <hip/hip_bf16.h>

typedef unsigned short ushort_t;
typedef __attribute__((ext_vector_type(16))) __bf16 v16bf;
typedef __attribute__((ext_vector_type(8)))  float  v8f;

union FragU { uint4 q[2]; v16bf v; };

// ---------- helpers ----------
static __device__ __forceinline__ ushort_t f2bf(float f) {
    union { float f; unsigned u; } v; v.f = f;
    unsigned r = v.u + 0x7FFFu + ((v.u >> 16) & 1u);
    return (ushort_t)(r >> 16);
}
static __device__ __forceinline__ float bf2f(ushort_t u) {
    union { unsigned u; float f; } v; v.u = ((unsigned)u) << 16;
    return v.f;
}

// ---------- problem constants ----------
#define BROWS      8192
#define IN_DIM     784
#define IN_PAD     800           // 784 padded to multiple of 32
#define DIM_I      4096
#define DIM_D      8192
#define K_NEURONS  512
#define NSTRIPES   128
#define SDIM       64
#define K_STRIPES  16

// =====================================================================
// cast float -> bf16 with column zero-padding (row-major [rows x Kd])
// =====================================================================
__global__ __launch_bounds__(256) void cast_pad_kernel(
    const float* __restrict__ src, ushort_t* __restrict__ dst,
    int rows, int Ks, int Kd)
{
    int idx = blockIdx.x * 256 + threadIdx.x;
    int total = rows * Kd;
    if (idx >= total) return;
    int r = idx / Kd, k = idx - r * Kd;
    float v = (k < Ks) ? src[(size_t)r * Ks + k] : 0.0f;
    dst[idx] = f2bf(v);
}

// =====================================================================
// transpose + cast: W [K x N] fp32 -> Wt [N x Kpad] bf16 (zero pad K)
// =====================================================================
__global__ __launch_bounds__(256) void transpose_cast_kernel(
    const float* __restrict__ W, ushort_t* __restrict__ Wt,
    int K, int N, int Kpad)
{
    int idx = blockIdx.x * 256 + threadIdx.x;
    int total = N * Kpad;
    if (idx >= total) return;
    int n = idx / Kpad, k = idx - n * Kpad;
    float v = (k < K) ? W[(size_t)k * N + n] : 0.0f;
    Wt[idx] = f2bf(v);
}

// =====================================================================
// WMMA bf16 GEMM: C[M,N] = relu(A[M,K] * Bt[N,K]^T + bias)
// 256 threads, 8 waves. Tile 128x128, BK=32, DOUBLE-BUFFERED LDS.
// One barrier per K-step; next tile's global loads overlap the WMMAs.
// Wave tile 64x32 (4x2 wmma accumulators).
// =====================================================================
#define AS_STRIDE 40   // LDS row stride in halfs (padded vs 32)

template<bool STORE_BF16, bool GUARD_N>
__global__ __launch_bounds__(256) void gemm_wmma_kernel(
    const ushort_t* __restrict__ A, const ushort_t* __restrict__ Bt,
    const float* __restrict__ bias, void* __restrict__ Cout,
    int M, int N, int K, int ldc)
{
    __shared__ ushort_t As[2][128 * AS_STRIDE];
    __shared__ ushort_t Bs[2][128 * AS_STRIDE];

    const int tid  = threadIdx.x;
    const int wave = tid >> 5;
    const int lane = tid & 31;
    const int wr   = wave >> 2;     // 0..1 -> 64-row band
    const int wc   = wave & 3;      // 0..3 -> 32-col band
    const int m0   = blockIdx.y * 128;
    const int n0   = blockIdx.x * 128;
    const int g    = lane >> 4;     // lane half
    const int r    = lane & 15;

    // staging coordinates: each thread owns two (row, 8-half seg) chunks
    const int row0 = tid >> 2;            // 0..63
    const int row1 = row0 + 64;           // 64..127
    const int seg  = (tid & 3) * 8;       // half offset within 32-wide tile

    const size_t aBase0 = (size_t)(m0 + row0) * K + seg;
    const size_t aBase1 = (size_t)(m0 + row1) * K + seg;
    const size_t bBase0 = (size_t)(n0 + row0) * K + seg;
    const size_t bBase1 = (size_t)(n0 + row1) * K + seg;
    const bool   bOk0   = (!GUARD_N) || (n0 + row0) < N;
    const bool   bOk1   = (!GUARD_N) || (n0 + row1) < N;

    v8f acc[4][2] = {};

    const int nsteps = K / 32;
    uint4 va0, va1, vb0, vb1;

    // ---- prologue: stage tile 0 into buffer 0
    va0 = *(const uint4*)(A + aBase0);
    va1 = *(const uint4*)(A + aBase1);
    vb0 = bOk0 ? *(const uint4*)(Bt + bBase0) : make_uint4(0u, 0u, 0u, 0u);
    vb1 = bOk1 ? *(const uint4*)(Bt + bBase1) : make_uint4(0u, 0u, 0u, 0u);
    *(uint4*)(&As[0][row0 * AS_STRIDE + seg]) = va0;
    *(uint4*)(&As[0][row1 * AS_STRIDE + seg]) = va1;
    *(uint4*)(&Bs[0][row0 * AS_STRIDE + seg]) = vb0;
    *(uint4*)(&Bs[0][row1 * AS_STRIDE + seg]) = vb1;
    __syncthreads();

    for (int s = 0; s < nsteps; ++s) {
        const int cur = s & 1;
        const int nxt = cur ^ 1;
        const bool have_next = (s + 1) < nsteps;

        // ---- issue next tile's global loads early (overlap with WMMA)
        if (have_next) {
            const int k0 = (s + 1) * 32;
            va0 = *(const uint4*)(A + aBase0 + k0);
            va1 = *(const uint4*)(A + aBase1 + k0);
            vb0 = bOk0 ? *(const uint4*)(Bt + bBase0 + k0) : make_uint4(0u, 0u, 0u, 0u);
            vb1 = bOk1 ? *(const uint4*)(Bt + bBase1 + k0) : make_uint4(0u, 0u, 0u, 0u);
            if ((s + 2) < nsteps) {
                __builtin_prefetch(A + aBase0 + (s + 2) * 32, 0, 1);
                __builtin_prefetch(Bt + bBase0 + (s + 2) * 32, 0, 1);
            }
        }

        // ---- fragments from current buffer
        v16bf afrag[4], bfrag[2];
#pragma unroll
        for (int mi = 0; mi < 4; ++mi) {
            // 16-bit A 16x32: lane half g -> K groups {g*8..g*8+7, 16+g*8..}
            const ushort_t* p = &As[cur][(wr * 64 + mi * 16 + r) * AS_STRIDE + g * 8];
            FragU u;
            u.q[0] = *(const uint4*)(p);
            u.q[1] = *(const uint4*)(p + 16);
            afrag[mi] = u.v;
        }
#pragma unroll
        for (int ni = 0; ni < 2; ++ni) {
            // 16-bit B 32x16: lane half g -> K groups {g*8..g*8+7, 16+g*8..}
            const ushort_t* p = &Bs[cur][(wc * 32 + ni * 16 + r) * AS_STRIDE + g * 8];
            FragU u;
            u.q[0] = *(const uint4*)(p);
            u.q[1] = *(const uint4*)(p + 16);
            bfrag[ni] = u.v;
        }

        // ---- 8 WMMA ops
#pragma unroll
        for (int mi = 0; mi < 4; ++mi)
#pragma unroll
            for (int ni = 0; ni < 2; ++ni)
                acc[mi][ni] = __builtin_amdgcn_wmma_f32_16x16x32_bf16(
                    false, afrag[mi], false, bfrag[ni],
                    (short)0, acc[mi][ni], false, false);

        // ---- commit next tile to the other buffer, single barrier
        if (have_next) {
            *(uint4*)(&As[nxt][row0 * AS_STRIDE + seg]) = va0;
            *(uint4*)(&As[nxt][row1 * AS_STRIDE + seg]) = va1;
            *(uint4*)(&Bs[nxt][row0 * AS_STRIDE + seg]) = vb0;
            *(uint4*)(&Bs[nxt][row1 * AS_STRIDE + seg]) = vb1;
        }
        __syncthreads();
    }

    // ---- epilogue: bias + relu, store (C layout: VGPR j -> M = j + 8*g, col = r)
#pragma unroll
    for (int ni = 0; ni < 2; ++ni) {
        int col = n0 + wc * 32 + ni * 16 + r;
        bool colok = (!GUARD_N) || (col < N);
        float bv = colok ? bias[col] : 0.0f;
#pragma unroll
        for (int mi = 0; mi < 4; ++mi) {
#pragma unroll
            for (int j = 0; j < 8; ++j) {
                int row = m0 + wr * 64 + mi * 16 + g * 8 + j;
                float v = acc[mi][ni][j] + bv;
                v = fmaxf(v, 0.0f);
                if (colok) {
                    if (STORE_BF16)
                        ((ushort_t*)Cout)[(size_t)row * ldc + col] = f2bf(v);
                    else
                        ((float*)Cout)[(size_t)row * ldc + col] = v;
                }
            }
        }
    }
}

// =====================================================================
// per-row top-k threshold via 2-level radix select on bf16 bit patterns
// (valid since h2 >= 0 after relu -> uint16 order == float order)
// =====================================================================
__global__ __launch_bounds__(256) void row_topk_threshold_kernel(
    const ushort_t* __restrict__ h2, ushort_t* __restrict__ thr, int D, int k)
{
    __shared__ unsigned hist[256];
    __shared__ unsigned s_hi, s_rem;
    const int row = blockIdx.x;
    const int tid = threadIdx.x;
    const ushort_t* p = h2 + (size_t)row * D;

    hist[tid] = 0;
    __syncthreads();
    for (int d = tid; d < D; d += 256)
        atomicAdd(&hist[p[d] >> 8], 1u);
    __syncthreads();
    if (tid == 0) {
        unsigned cum = 0; unsigned b = 0, rem = (unsigned)k;
        for (int i = 255; i >= 0; --i) {
            if (cum + hist[i] >= (unsigned)k) { b = (unsigned)i; rem = (unsigned)k - cum; break; }
            cum += hist[i];
        }
        s_hi = b; s_rem = rem;
    }
    __syncthreads();
    unsigned hb = s_hi, rem = s_rem;
    hist[tid] = 0;
    __syncthreads();
    for (int d = tid; d < D; d += 256) {
        unsigned u = p[d];
        if ((u >> 8) == hb) atomicAdd(&hist[u & 255u], 1u);
    }
    __syncthreads();
    if (tid == 0) {
        unsigned cum = 0; unsigned lb = 0;
        for (int i = 255; i >= 0; --i) {
            if (cum + hist[i] >= rem) { lb = (unsigned)i; break; }
            cum += hist[i];
        }
        thr[row] = (ushort_t)((hb << 8) | lb);
    }
}

// =====================================================================
// apply neuron mask + boosts + square, stripe top-16 mask, emit bf16 code
// one block (256 thr) per row; each thread owns 32 contiguous elements
// =====================================================================
__global__ __launch_bounds__(256) void topk_apply_kernel(
    const ushort_t* __restrict__ h2, const ushort_t* __restrict__ thr,
    const float* __restrict__ bscore, ushort_t* __restrict__ codebf, int D)
{
    __shared__ float psum[256];
    __shared__ float savg[NSTRIPES];
    __shared__ unsigned char keep[NSTRIPES];
    const int row = blockIdx.x;
    const int tid = threadIdx.x;
    const ushort_t* p = h2 + (size_t)row * D;
    const ushort_t t = thr[row];
    const int CH = D / 256;            // 32 elems/thread -> half a stripe
    const int base = tid * CH;

    float vals[32];
    float local = 0.0f;
#pragma unroll 4
    for (int i = 0; i < CH; ++i) {
        ushort_t u = p[base + i];
        float v = bf2f(u);
        float boost = __expf(-bscore[base + i]);   // exp(BETA*(GAMMA - s)), BETA=1, GAMMA=0
        float c = (u >= t) ? boost * v * v : 0.0f;
        vals[i] = c;
        local += c;
    }
    psum[tid] = local;
    __syncthreads();
    if (tid < NSTRIPES) savg[tid] = psum[2 * tid] + psum[2 * tid + 1];
    __syncthreads();
    if (tid < NSTRIPES) {
        float me = savg[tid];
        int cnt = 0;
#pragma unroll 8
        for (int j = 0; j < NSTRIPES; ++j) {
            float o = savg[j];
            cnt += (o > me) || (o == me && j < tid);
        }
        keep[tid] = (cnt < K_STRIPES) ? 1 : 0;
    }
    __syncthreads();
    const bool kp = keep[tid >> 1] != 0;
    ushort_t* q = codebf + (size_t)row * D + base;
#pragma unroll 4
    for (int i = 0; i < CH; ++i)
        q[i] = kp ? f2bf(vals[i]) : (ushort_t)0;
}

// =====================================================================
// launch
// =====================================================================
static inline size_t align256(size_t x) { return (x + 255) & ~(size_t)255; }

extern "C" void kernel_launch(void* const* d_in, const int* in_sizes, int n_in,
                              void* d_out, int out_size, void* d_ws, size_t ws_size,
                              hipStream_t stream) {
    (void)in_sizes; (void)n_in; (void)out_size; (void)ws_size;

    const float* x  = (const float*)d_in[0];
    const float* W1 = (const float*)d_in[1];
    const float* b1 = (const float*)d_in[2];
    const float* W2 = (const float*)d_in[3];
    const float* b2 = (const float*)d_in[4];
    const float* W3 = (const float*)d_in[5];
    const float* b3 = (const float*)d_in[6];
    const float* W4 = (const float*)d_in[7];
    const float* b4 = (const float*)d_in[8];
    const float* bs = (const float*)d_in[9];
    float* out = (float*)d_out;

    char* ws = (char*)d_ws;
    size_t ofs = 0;
    ushort_t* xbf  = (ushort_t*)(ws + ofs); ofs += align256((size_t)BROWS * IN_PAD * 2);
    ushort_t* W1t  = (ushort_t*)(ws + ofs); ofs += align256((size_t)DIM_I * IN_PAD * 2);
    ushort_t* W2t  = (ushort_t*)(ws + ofs); ofs += align256((size_t)DIM_D * DIM_I * 2);
    ushort_t* W3t  = (ushort_t*)(ws + ofs); ofs += align256((size_t)DIM_I * DIM_D * 2);
    ushort_t* W4t  = (ushort_t*)(ws + ofs); ofs += align256((size_t)IN_DIM * DIM_I * 2);
    ushort_t* h1bf = (ushort_t*)(ws + ofs); ofs += align256((size_t)BROWS * DIM_I * 2);
    ushort_t* h2bf = (ushort_t*)(ws + ofs); ofs += align256((size_t)BROWS * DIM_D * 2);
    ushort_t* code = (ushort_t*)(ws + ofs); ofs += align256((size_t)BROWS * DIM_D * 2);
    ushort_t* h3bf = (ushort_t*)(ws + ofs); ofs += align256((size_t)BROWS * DIM_I * 2);
    ushort_t* thr  = (ushort_t*)(ws + ofs); ofs += align256((size_t)BROWS * 2);

    // ---- casts / transposes
    {
        int total = BROWS * IN_PAD;
        cast_pad_kernel<<<(total + 255) / 256, 256, 0, stream>>>(x, xbf, BROWS, IN_DIM, IN_PAD);
    }
    {
        int total = DIM_I * IN_PAD;
        transpose_cast_kernel<<<(total + 255) / 256, 256, 0, stream>>>(W1, W1t, IN_DIM, DIM_I, IN_PAD);
    }
    {
        int total = DIM_D * DIM_I;
        transpose_cast_kernel<<<(total + 255) / 256, 256, 0, stream>>>(W2, W2t, DIM_I, DIM_D, DIM_I);
    }
    {
        int total = DIM_I * DIM_D;
        transpose_cast_kernel<<<(total + 255) / 256, 256, 0, stream>>>(W3, W3t, DIM_D, DIM_I, DIM_D);
    }
    {
        int total = IN_DIM * DIM_I;
        transpose_cast_kernel<<<(total + 255) / 256, 256, 0, stream>>>(W4, W4t, DIM_I, IN_DIM, DIM_I);
    }

    // ---- GEMM1: h1 = relu(x @ W1 + b1)   [8192 x 4096], K=800
    {
        dim3 grid(DIM_I / 128, BROWS / 128);
        gemm_wmma_kernel<true, false><<<grid, 256, 0, stream>>>(
            xbf, W1t, b1, (void*)h1bf, BROWS, DIM_I, IN_PAD, DIM_I);
    }
    // ---- GEMM2: h2 = relu(h1 @ W2 + b2)  [8192 x 8192], K=4096
    {
        dim3 grid(DIM_D / 128, BROWS / 128);
        gemm_wmma_kernel<true, false><<<grid, 256, 0, stream>>>(
            h1bf, W2t, b2, (void*)h2bf, BROWS, DIM_D, DIM_I, DIM_D);
    }
    // ---- top-k neuron threshold + stripe mask -> code
    row_topk_threshold_kernel<<<BROWS, 256, 0, stream>>>(h2bf, thr, DIM_D, K_NEURONS);
    topk_apply_kernel<<<BROWS, 256, 0, stream>>>(h2bf, thr, bs, code, DIM_D);

    // ---- GEMM3: h3 = relu(code @ W3 + b3) [8192 x 4096], K=8192
    {
        dim3 grid(DIM_I / 128, BROWS / 128);
        gemm_wmma_kernel<true, false><<<grid, 256, 0, stream>>>(
            code, W3t, b3, (void*)h3bf, BROWS, DIM_I, DIM_D, DIM_I);
    }
    // ---- GEMM4: out = relu(h3 @ W4 + b4)  [8192 x 784], K=4096, guarded N
    {
        dim3 grid((IN_DIM + 127) / 128, BROWS / 128);
        gemm_wmma_kernel<false, true><<<grid, 256, 0, stream>>>(
            h3bf, W4t, b4, (void*)out, BROWS, IN_DIM, DIM_I, IN_DIM);
    }
}
